// Conv_DCNv4_83021717831893
// MI455X (gfx1250) — compile-verified, matrix operand order checked by reading, not run on hardware
//
#include <hip/hip_runtime.h>
#include <hip/hip_bf16.h>
#include <math.h>

// ---------------- problem constants ----------------
constexpr int kC   = 256;
constexpr int kH   = 128;
constexpr int kW   = 128;
constexpr int kHW  = kH * kW;          // 16384
constexpr int kM   = 4 * kHW;          // 65536 spatial positions
constexpr int kG   = 4;
constexpr int kCg  = kC / kG;          // 64
constexpr int kPad = 112;              // PAD_OFF
constexpr float kEps = 1e-5f;

// ---------------- WMMA fragment types ----------------
typedef __attribute__((ext_vector_type(16))) __bf16 v16bf;
typedef __attribute__((ext_vector_type(8)))  float  v8f;

union AFrag {
    v16bf v;
    uint4 q[2];       // two contiguous 16B runs per lane (CDNA5 16-bit layout)
};

__device__ __forceinline__ unsigned short f2bf(float f) {
    unsigned u = __builtin_bit_cast(unsigned, f);
    u += 0x7FFFu + ((u >> 16) & 1u);   // round-to-nearest-even
    return (unsigned short)(u >> 16);
}
__device__ __forceinline__ unsigned packbf2(float a, float b) {
    return (unsigned)f2bf(a) | ((unsigned)f2bf(b) << 16);
}

// ---------------------------------------------------------------------------
// One-time fp32 -> bf16 conversion (weights)
// ---------------------------------------------------------------------------
__global__ __launch_bounds__(256) void cvt_bf16_kernel(
    const float* __restrict__ src, unsigned short* __restrict__ dst, int n)
{
    const int i = blockIdx.x * 256 + threadIdx.x;
    if (i < n) dst[i] = f2bf(src[i]);
}

// ---------------------------------------------------------------------------
// Fused GEMM + epilogue.   D[m][n] = sum_k A[m][k] * W[n][k]   (K = 256)
// ALAYOUT: 0 -> A fp32 NCHW (A[m][k] = x[img][k][hw], contiguous in m)
//          1 -> A bf16 row-major M x 256 (contiguous in k)
// EPI:     0 -> + bias, fp32 row-major store (ldd)
//          1 -> BN1 + SiLU, bf16 row-major store (t)
//          2 -> + bias, BN2 + SiLU, fp32 NCHW store (d_out)
// Block = 128 threads (4 waves). Block tile = 128(M) x 16(N).
// Each wave: rows [wid*32, wid*32+32) -> two 16x16 accumulators sharing one
// B fragment; K stepped by 32 -> 2 x v_wmma_f32_16x16x32_bf16 per step.
// ---------------------------------------------------------------------------
template <int ALAYOUT, int EPI>
__global__ __launch_bounds__(128) void gemm_epi_kernel(
    const void* __restrict__ Ap, const unsigned short* __restrict__ Wbf,
    const float* __restrict__ bias,
    const float* __restrict__ gam, const float* __restrict__ bet,
    const float* __restrict__ mu,  const float* __restrict__ var,
    void* __restrict__ Dp, int ldd)
{
    // 128 rows x 32 k bf16, row stride 40 ushorts (80B, 16B-aligned fragments)
    __shared__ unsigned short sA[128 * 40];

    const int tid  = threadIdx.x;
    const int wid  = tid >> 5;
    const int lane = tid & 31;
    const int mr   = lane & 15;    // row within 16 (A) / col within 16 (B,D)
    const int kh   = lane >> 4;    // K-half select per ISA 16-bit layout
    const int m0   = blockIdx.x * 128;
    const int ncol = blockIdx.y * 16 + mr;

    const float*          Af = (const float*)Ap;
    const unsigned short* Ab = (const unsigned short*)Ap;
    const unsigned short* Wrow = Wbf + (size_t)ncol * kC;

    const int img = m0 >> 14;          // 128 | 16384 -> tiles never straddle images
    const int hw0 = m0 & (kHW - 1);

    v8f acc0 = {};
    v8f acc1 = {};

    for (int k0 = 0; k0 < kC; k0 += 32) {
        // ---- stage the 128x32 A tile into LDS ----
        if (ALAYOUT == 0) {
            // fp32 NCHW: vectorize along contiguous m (float4), convert to bf16
            #pragma unroll
            for (int e = 0; e < 8; ++e) {
                const int idx = e * 128 + tid;        // 1024 float4 chunks
                const int m   = (idx & 31) * 4;
                const int k   = idx >> 5;
                const float4 f = *(const float4*)(
                    Af + (size_t)img * (kC * kHW) + (size_t)(k0 + k) * kHW + (hw0 + m));
                sA[(m + 0) * 40 + k] = f2bf(f.x);
                sA[(m + 1) * 40 + k] = f2bf(f.y);
                sA[(m + 2) * 40 + k] = f2bf(f.z);
                sA[(m + 3) * 40 + k] = f2bf(f.w);
            }
        } else {
            // bf16 row-major: pure b128 copies, no conversion
            #pragma unroll
            for (int e = 0; e < 4; ++e) {
                const int idx = e * 128 + tid;        // 512 uint4 chunks
                const int m   = idx >> 2;
                const int kc  = (idx & 3) * 8;
                const uint4 d = *(const uint4*)(Ab + (size_t)(m0 + m) * kC + k0 + kc);
                *(uint4*)&sA[m * 40 + kc] = d;
            }
            if (k0 + 32 < kC)
                __builtin_prefetch(Ab + (size_t)(m0 + tid) * kC + k0 + 32, 0, 3);
        }
        __syncthreads();

        // ---- fragments per CDNA5 16-bit A/B layout ----
        AFrag a0, a1, b;
        const int krow = kh * 8;                       // element offset of K-half
        a0.q[0] = *(const uint4*)&sA[(wid * 32 + mr) * 40 + krow];
        a0.q[1] = *(const uint4*)&sA[(wid * 32 + mr) * 40 + 16 + krow];
        a1.q[0] = *(const uint4*)&sA[(wid * 32 + 16 + mr) * 40 + krow];
        a1.q[1] = *(const uint4*)&sA[(wid * 32 + 16 + mr) * 40 + 16 + krow];
        b.q[0]  = *(const uint4*)(Wrow + k0 + krow);
        b.q[1]  = *(const uint4*)(Wrow + k0 + 16 + krow);

        acc0 = __builtin_amdgcn_wmma_f32_16x16x32_bf16(
                   false, a0.v, false, b.v, (short)0, acc0, false, false);
        acc1 = __builtin_amdgcn_wmma_f32_16x16x32_bf16(
                   false, a1.v, false, b.v, (short)0, acc1, false, false);
        __syncthreads();
    }

    // ---- epilogue: lane owns column ncol; frag0 rows base0+r, frag1 +16 ----
    const int base0 = m0 + wid * 32 + kh * 8;
    if (EPI == 0) {
        float* D = (float*)Dp;
        const float bs = bias[ncol];
        #pragma unroll
        for (int r = 0; r < 8; ++r) {
            D[(size_t)(base0 + r)      * ldd + ncol] = acc0[r] + bs;
            D[(size_t)(base0 + 16 + r) * ldd + ncol] = acc1[r] + bs;
        }
    } else if (EPI == 1) {
        unsigned short* D = (unsigned short*)Dp;
        const float sc = gam[ncol] * rsqrtf(var[ncol] + kEps);
        const float sh = bet[ncol] - mu[ncol] * sc;
        #pragma unroll
        for (int r = 0; r < 8; ++r) {
            float y0 = acc0[r] * sc + sh;
            float y1 = acc1[r] * sc + sh;
            y0 = y0 / (1.0f + __expf(-y0));            // SiLU
            y1 = y1 / (1.0f + __expf(-y1));
            D[(size_t)(base0 + r)      * kC + ncol] = f2bf(y0);
            D[(size_t)(base0 + 16 + r) * kC + ncol] = f2bf(y1);
        }
    } else {
        float* D = (float*)Dp;
        const float bs = bias[ncol];
        const float sc = gam[ncol] * rsqrtf(var[ncol] + kEps);
        const float sh = bet[ncol] - mu[ncol] * sc;
        #pragma unroll
        for (int r = 0; r < 8; ++r) {
            float y0 = (acc0[r] + bs) * sc + sh;
            float y1 = (acc1[r] + bs) * sc + sh;
            y0 = y0 / (1.0f + __expf(-y0));            // SiLU
            y1 = y1 / (1.0f + __expf(-y1));
            const int ma = base0 + r;
            const int mb = base0 + 16 + r;
            D[(size_t)(ma >> 14) * (kC * kHW) + (size_t)ncol * kHW + (ma & (kHW - 1))] = y0;
            D[(size_t)(mb >> 14) * (kC * kHW) + (size_t)ncol * kHW + (mb & (kHW - 1))] = y1;
        }
    }
}

// ---------------------------------------------------------------------------
// Deformable bilinear sampling.
// One wave per (position p, group g); 32 lanes split Cg=64 channels (2 each,
// float2 loads -> fully coalesced 256B per bilinear corner).
// v fp32 [p][g*64+cg]; om fp32 [p][112]; output bf16 [p][g*64+cg].
// ---------------------------------------------------------------------------
__global__ __launch_bounds__(256) void dcn_sample_kernel(
    const float* __restrict__ v, const float* __restrict__ om,
    unsigned short* __restrict__ outb)
{
    const int tid  = threadIdx.x;
    const int wid  = tid >> 5;
    const int lane = tid & 31;
    const int pair = blockIdx.x * 8 + wid;   // (p,g) pair index
    const int p    = pair >> 2;
    const int g    = pair & 3;
    const int img  = p >> 14;
    const int hw   = p & (kHW - 1);
    const int hc   = hw >> 7;
    const int wc   = hw & (kW - 1);

    const float* omg = om + (size_t)p * kPad + g * 27;   // 18 offsets + 9 masks
    const int    cg  = lane * 2;
    const float* vg  = v + g * kCg + cg;
    const size_t imgbase = (size_t)img * kHW;

    float o0 = 0.f, o1 = 0.f;
    #pragma unroll
    for (int k = 0; k < 9; ++k) {
        const int di = k / 3 - 1;
        const int dj = k % 3 - 1;
        const float lh = (float)(hc + di) + omg[2 * k + 1];
        const float lw = (float)(wc + dj) + omg[2 * k + 0];
        const float mk = omg[18 + k];
        const float fh = floorf(lh), fw = floorf(lw);
        const float dh = lh - fh,   dw = lw - fw;
        const int   h0 = (int)fh,   w0 = (int)fw;

        float a0 = 0.f, a1 = 0.f;
        #pragma unroll
        for (int c = 0; c < 4; ++c) {
            const int hh2 = h0 + (c >> 1);
            const int ww2 = w0 + (c & 1);
            const float wt = ((c >> 1) ? dh : 1.f - dh) * ((c & 1) ? dw : 1.f - dw);
            if (hh2 >= 0 && hh2 < kH && ww2 >= 0 && ww2 < kW) {
                const float2 val =
                    *(const float2*)(vg + (imgbase + (size_t)(hh2 * kW + ww2)) * kC);
                a0 += wt * val.x;
                a1 += wt * val.y;
            }
        }
        o0 += mk * a0;
        o1 += mk * a1;
    }
    *(unsigned*)(outb + (size_t)p * kC + g * kCg + cg) = packbf2(o0, o1);
}

// ---------------------------------------------------------------------------
extern "C" void kernel_launch(void* const* d_in, const int* in_sizes, int n_in,
                              void* d_out, int out_size, void* d_ws, size_t ws_size,
                              hipStream_t stream) {
    (void)in_sizes; (void)n_in; (void)out_size; (void)ws_size;
    const float* x       = (const float*)d_in[0];
    const float* conv_w  = (const float*)d_in[1];   // (C,C,1,1) == [o][c]
    const float* bn1g    = (const float*)d_in[2];
    const float* bn1b    = (const float*)d_in[3];
    const float* bn1m    = (const float*)d_in[4];
    const float* bn1v    = (const float*)d_in[5];
    const float* value_w = (const float*)d_in[6];
    const float* value_b = (const float*)d_in[7];
    const float* off_w   = (const float*)d_in[8];
    const float* off_b   = (const float*)d_in[9];
    const float* out_w   = (const float*)d_in[10];
    const float* out_b   = (const float*)d_in[11];
    const float* bn2g    = (const float*)d_in[12];
    const float* bn2b    = (const float*)d_in[13];
    const float* bn2m    = (const float*)d_in[14];
    const float* bn2v    = (const float*)d_in[15];
    float* out = (float*)d_out;

    // workspace layout (bytes):
    //   t_bf   bf16 [65536][256]  32MB   (reused as "sampled" after stage 3)
    //   v      fp32 [65536][256]  64MB
    //   om     fp32 [65536][112]  28MB
    //   bf16 weights: conv | value | offset | out
    char* ws = (char*)d_ws;
    unsigned short* t_bf = (unsigned short*)ws;
    float*          v    = (float*)(ws + (size_t)33554432);
    float*          om   = (float*)(ws + (size_t)33554432 + 67108864);
    unsigned short* wc   = (unsigned short*)(ws + (size_t)33554432 + 67108864 + 29360128);
    unsigned short* wv   = wc + 65536;
    unsigned short* wo   = wv + 65536;
    unsigned short* wu   = wo + 28672;

    // 0) weights fp32 -> bf16 (one-time per call, tiny)
    cvt_bf16_kernel<<<dim3(256), dim3(256), 0, stream>>>(conv_w,  wc, 65536);
    cvt_bf16_kernel<<<dim3(256), dim3(256), 0, stream>>>(value_w, wv, 65536);
    cvt_bf16_kernel<<<dim3(112), dim3(256), 0, stream>>>(off_w,   wo, 28672);
    cvt_bf16_kernel<<<dim3(256), dim3(256), 0, stream>>>(out_w,   wu, 65536);

    const dim3 blk(128);

    // 1) 1x1 conv + BN1 + SiLU -> t (bf16 NHWC row-major)
    gemm_epi_kernel<0, 1><<<dim3(kM / 128, kC / 16), blk, 0, stream>>>(
        x, wc, nullptr, bn1g, bn1b, bn1m, bn1v, t_bf, kC);

    // 2a) value projection -> v (fp32)
    gemm_epi_kernel<1, 0><<<dim3(kM / 128, kC / 16), blk, 0, stream>>>(
        t_bf, wv, value_b, nullptr, nullptr, nullptr, nullptr, v, kC);

    // 2b) offset/mask projection -> om (fp32, N = 112)
    gemm_epi_kernel<1, 0><<<dim3(kM / 128, kPad / 16), blk, 0, stream>>>(
        t_bf, wo, off_b, nullptr, nullptr, nullptr, nullptr, om, kPad);

    // 3) deformable bilinear gather -> reuse t buffer as bf16 "sampled"
    dcn_sample_kernel<<<dim3(kM * kG / 8), dim3(256), 0, stream>>>(v, om, t_bf);

    // 4) output projection + BN2 + SiLU -> d_out (fp32 NCHW)
    gemm_epi_kernel<1, 2><<<dim3(kM / 128, kC / 16), blk, 0, stream>>>(
        t_bf, wu, out_b, bn2g, bn2b, bn2m, bn2v, out, kC);
}
